// OnsiteModel_16544214024457
// MI455X (gfx1250) — matrix operand order repaired and checked by reading, not compiled, not run on hardware
//
#include <hip/hip_runtime.h>
#include <math.h>

// ---------------------------------------------------------------------------
// Types matching the gfx1250 WMMA builtin signatures
// ---------------------------------------------------------------------------
typedef __attribute__((ext_vector_type(16))) __bf16 v16bf;
typedef __attribute__((ext_vector_type(8)))  float  v8f;

// ---------------------------------------------------------------------------
// Packed bf16 weight buffer layout (element offsets inside d_ws)
// Each weight is stored as WMMA-B fragments:
//   frag index = (chunk*ntiles + tile), 32 lanes * 16 bf16 per frag.
//   lane L: column = tile*16 + (L&15); element j (0..15): K = chunk*32 + (L>>4)*8 + j + (j>>3)*8
// ---------------------------------------------------------------------------
#define O_W000   0
#define O_W110   (O_W000  + 4096*128)
#define O_W011   (O_W110  + 1024*128)
#define O_W112   (O_W011  + 2048*64)
#define O_W2000  (O_W112  + 1024*32)
#define O_W2110  (O_W2000 + 32*32)
#define O_W2220  (O_W2110 + 64*64)

__global__ __launch_bounds__(256) void pack_weight(const float* __restrict__ src,
                                                   __bf16* __restrict__ dst,
                                                   int K, int Nw, float scale) {
    int nt = Nw >> 4;
    long total = (long)K * (long)Nw;
    for (long g = (long)blockIdx.x * blockDim.x + threadIdx.x; g < total;
         g += (long)gridDim.x * blockDim.x) {
        int j    = (int)(g & 15);
        int lane = (int)((g >> 4) & 31);
        long ct  = g >> 9;
        int t    = (int)(ct % nt);
        int c    = (int)(ct / nt);
        int k    = c * 32 + ((lane >> 4) << 3) + j + ((j >> 3) << 3);
        int col  = t * 16 + (lane & 15);
        dst[g] = (__bf16)(src[(long)k * Nw + col] * scale);
    }
}

// ---------------------------------------------------------------------------
// Device helpers
// ---------------------------------------------------------------------------
__device__ __forceinline__ v8f v8f_zero() {
    v8f z;
#pragma unroll
    for (int i = 0; i < 8; ++i) z[i] = 0.0f;
    return z;
}

__device__ __forceinline__ v8f wmma_bf16(v16bf a, v16bf b, v8f c) {
    return __builtin_amdgcn_wmma_f32_16x16x32_bf16(false, a, false, b, (short)0, c,
                                                   false, false);
}

// Load one pre-swizzled B fragment: two 16B reads per lane.
__device__ __forceinline__ v16bf load_bfrag(const __bf16* w, int c, int nt, int t,
                                            int lane) {
    const uint4* p = (const uint4*)(w + ((size_t)((c * nt + t) * 32 + lane) << 4));
    union { uint4 u[2]; v16bf v; } un;
    un.u[0] = p[0];
    un.u[1] = p[1];
    return un.v;
}

// A fragment = scalar * vector (16-bit A layout: K = kb..kb+7 and 16+kb..16+kb+7)
__device__ __forceinline__ v16bf afrag_scale(float s, const float* pv) {
    v16bf a;
#pragma unroll
    for (int j = 0; j < 8; ++j) {
        a[j]     = (__bf16)(s * pv[j]);
        a[8 + j] = (__bf16)(s * pv[16 + j]);
    }
    return a;
}

// A fragment = 3-term dot (for the x1.x1 -> 0e channel)
__device__ __forceinline__ v16bf afrag_dot3(float s0, float s1, float s2,
                                            const float* p0, const float* p1,
                                            const float* p2) {
    v16bf a;
#pragma unroll
    for (int j = 0; j < 8; ++j) {
        a[j]     = (__bf16)(s0 * p0[j] + s1 * p1[j] + s2 * p2[j]);
        a[8 + j] = (__bf16)(s0 * p0[16 + j] + s1 * p1[16 + j] + s2 * p2[16 + j]);
    }
    return a;
}

// Per-wave LDS store->load ordering (DS ops are in-order within a wave).
__device__ __forceinline__ void lds_fence() {
    __builtin_amdgcn_wave_barrier();
    asm volatile("s_wait_dscnt 0" ::: "memory");
}

// z contribution of a quadratic form  z[n] += sum_{u,v} Y[n,u] Y[n,v] W2[u,v]
template <int NT, int NC>
__device__ __forceinline__ void quad_form(const float* Y, int stride,
                                          const __bf16* w2, int lane, float* zacc) {
    const int ln15 = lane & 15;
    const int hf   = lane >> 4;
    const int kb   = hf * 8;
    v8f T[NT];
#pragma unroll
    for (int t = 0; t < NT; ++t) T[t] = v8f_zero();
#pragma unroll
    for (int c = 0; c < NC; ++c) {
        const float* p = Y + ln15 * stride + c * 32 + kb;
        v16bf a;
#pragma unroll
        for (int j = 0; j < 8; ++j) {
            a[j]     = (__bf16)p[j];
            a[8 + j] = (__bf16)p[16 + j];
        }
#pragma unroll
        for (int t = 0; t < NT; ++t)
            T[t] = wmma_bf16(a, load_bfrag(w2, c, NT, t, lane), T[t]);
    }
#pragma unroll
    for (int t = 0; t < NT; ++t) {
        const int col = t * 16 + ln15;
#pragma unroll
        for (int r = 0; r < 8; ++r)
            zacc[r] += T[t][r] * Y[(hf * 8 + r) * stride + col];
    }
}

// silu on cols 0..31, sigmoid on cols 32..127, store to act tile
__device__ __forceinline__ void act_store(const v8f* acc, float* A, int ln15, int hf) {
#pragma unroll
    for (int t = 0; t < 8; ++t) {
        const int col = t * 16 + ln15;
#pragma unroll
        for (int r = 0; r < 8; ++r) {
            const int m = hf * 8 + r;
            const float v  = acc[t][r];
            const float sg = 1.0f / (1.0f + __expf(-v));
            A[m * 128 + col] = (t < 2) ? (v * sg) : sg;
        }
    }
}

// multiply tile by gate column block from act and stash into ybuf
template <int NT>
__device__ __forceinline__ void gate_store(const v8f* T, const float* A, int gofs,
                                           float* Yb, int ln15, int hf) {
#pragma unroll
    for (int t = 0; t < NT; ++t) {
        const int col = t * 16 + ln15;
#pragma unroll
        for (int r = 0; r < 8; ++r) {
            const int m = hf * 8 + r;
            Yb[m * 64 + col] = T[t][r] * A[m * 128 + gofs + col];
        }
    }
}

// out2 (5 CG channels) + gate + readout for one 16-sample sub-tile
__device__ __forceinline__ void phase_out2(const float* X1r, const float* A, float* Yb,
                                           const __bf16* w112, const __bf16* w2220,
                                           int lane, int kb, int ln15, int hf,
                                           float* zacc) {
    const float INV_R2 = 0.70710678f;   // 1/sqrt(2)
    const float INV_R6 = 0.40824829f;   // 1/sqrt(6)
    v8f a2[5][2];
#pragma unroll
    for (int m = 0; m < 5; ++m)
#pragma unroll
        for (int t = 0; t < 2; ++t) a2[m][t] = v8f_zero();

    for (int c = 0; c < 32; ++c) {   // K = 1024
        const float xu0 = X1r[c], xu1 = X1r[32 + c], xu2 = X1r[64 + c];
        const float *p0 = X1r + kb, *p1 = X1r + 32 + kb, *p2 = X1r + 64 + kb;
        float e0[16], e1[16], e2[16];
#pragma unroll
        for (int j = 0; j < 8; ++j) {
            e0[j] = p0[j];  e0[8 + j] = p0[16 + j];
            e1[j] = p1[j];  e1[8 + j] = p1[16 + j];
            e2[j] = p2[j];  e2[8 + j] = p2[16 + j];
        }
        v16bf b0 = load_bfrag(w112, c, 2, 0, lane);
        v16bf b1 = load_bfrag(w112, c, 2, 1, lane);
        v16bf a;
#pragma unroll
        for (int j = 0; j < 16; ++j) a[j] = (__bf16)((xu0 * e1[j] + xu1 * e0[j]) * INV_R2);
        a2[0][0] = wmma_bf16(a, b0, a2[0][0]); a2[0][1] = wmma_bf16(a, b1, a2[0][1]);
#pragma unroll
        for (int j = 0; j < 16; ++j) a[j] = (__bf16)((xu1 * e2[j] + xu2 * e1[j]) * INV_R2);
        a2[1][0] = wmma_bf16(a, b0, a2[1][0]); a2[1][1] = wmma_bf16(a, b1, a2[1][1]);
#pragma unroll
        for (int j = 0; j < 16; ++j)
            a[j] = (__bf16)((2.0f * xu2 * e2[j] - xu0 * e0[j] - xu1 * e1[j]) * INV_R6);
        a2[2][0] = wmma_bf16(a, b0, a2[2][0]); a2[2][1] = wmma_bf16(a, b1, a2[2][1]);
#pragma unroll
        for (int j = 0; j < 16; ++j) a[j] = (__bf16)((xu0 * e2[j] + xu2 * e0[j]) * INV_R2);
        a2[3][0] = wmma_bf16(a, b0, a2[3][0]); a2[3][1] = wmma_bf16(a, b1, a2[3][1]);
#pragma unroll
        for (int j = 0; j < 16; ++j) a[j] = (__bf16)((xu0 * e0[j] - xu1 * e1[j]) * INV_R2);
        a2[4][0] = wmma_bf16(a, b0, a2[4][0]); a2[4][1] = wmma_bf16(a, b1, a2[4][1]);
    }
#pragma unroll 1
    for (int m = 0; m < 5; ++m) {
        gate_store<2>(a2[m], A, 96, Yb, ln15, hf);
        lds_fence();
        quad_form<2, 1>(Yb, 64, w2220, lane, zacc);   // z += y2^T W2_220 y2
        lds_fence();
    }
}

// ---------------------------------------------------------------------------
// Main fused kernel: 64 threads = 2 wave32; each wave owns 32 samples
// (two 16-row WMMA tiles sharing every B fragment).
// ---------------------------------------------------------------------------
__global__ __launch_bounds__(64) void tp_gate_tp_kernel(
    const float* __restrict__ x0g,   // [N,64]
    const float* __restrict__ x1g,   // [N,32,3]
    const __bf16* __restrict__ wp,   // packed weights
    float* __restrict__ out,         // [N]
    int nBlocks)                     // N/64
{
    __shared__ float x0s[64][64];        // [s][u]            16 KB
    __shared__ float x1s[64][96];        // [s][i*32+v]       24 KB
    __shared__ float act[2][2][16][128]; // [wave][tile]      32 KB
    __shared__ float ybuf[2][16][64];    // gated features     8 KB
    __shared__ float zpart[64];

    const __bf16* w000  = wp + O_W000;
    const __bf16* w110  = wp + O_W110;
    const __bf16* w011  = wp + O_W011;
    const __bf16* w112  = wp + O_W112;
    const __bf16* w2000 = wp + O_W2000;
    const __bf16* w2110 = wp + O_W2110;
    const __bf16* w2220 = wp + O_W2220;

    const int tid  = threadIdx.x;
    const int wv   = tid >> 5;
    const int lane = tid & 31;
    const int ln15 = lane & 15;
    const int hf   = lane >> 4;
    const int kb   = hf * 8;
    const int s0   = wv * 32;            // wave's first sample (owns 32)

    for (int blk = blockIdx.x; blk < nBlocks; blk += gridDim.x) {
        // ---- stage inputs (cooperative, whole WG) ----
        {
            const float4* src = (const float4*)(x0g + (size_t)blk * 64 * 64);
            float4* dst = (float4*)&x0s[0][0];
            for (int i = tid; i < 64 * 16; i += 64) dst[i] = src[i];
            for (int i = tid; i < 64 * 32; i += 64) {
                int s = i >> 5, v = i & 31;
                const float* p = x1g + ((size_t)(blk * 64 + s) * 32 + v) * 3;
                x1s[s][v]      = p[0];
                x1s[s][32 + v] = p[1];
                x1s[s][64 + v] = p[2];
            }
            zpart[tid] = 0.0f;
        }
        __syncthreads();

        const float* X0a = &x0s[s0 + ln15][0];
        const float* X0b = &x0s[s0 + 16 + ln15][0];
        const float* X1a = &x1s[s0 + ln15][0];
        const float* X1b = &x1s[s0 + 16 + ln15][0];
        float* Aact[2] = { &act[wv][0][0][0], &act[wv][1][0][0] };
        float* Yb = &ybuf[wv][0][0];
        float zacc[2][8];
#pragma unroll
        for (int st = 0; st < 2; ++st)
#pragma unroll
            for (int r = 0; r < 8; ++r) zacc[st][r] = 0.0f;

        // ===== Phase A: out0[32,128] = x0x0*W000 + x1x1*W110 =====
        v8f accA[8], accB[8];
#pragma unroll
        for (int t = 0; t < 8; ++t) { accA[t] = v8f_zero(); accB[t] = v8f_zero(); }

        for (int c = 0; c < 128; ++c) {          // G000, K = 4096
            const int u = c >> 1, v0 = (c & 1) << 5;
            v16bf aA = afrag_scale(X0a[u], X0a + v0 + kb);
            v16bf aB = afrag_scale(X0b[u], X0b + v0 + kb);
#pragma unroll
            for (int t = 0; t < 8; ++t) {
                v16bf b = load_bfrag(w000, c, 8, t, lane);
                accA[t] = wmma_bf16(aA, b, accA[t]);
                accB[t] = wmma_bf16(aB, b, accB[t]);
            }
        }
        for (int c = 0; c < 32; ++c) {           // G110, K = 1024
            v16bf aA = afrag_dot3(X1a[c], X1a[32 + c], X1a[64 + c],
                                  X1a + kb, X1a + 32 + kb, X1a + 64 + kb);
            v16bf aB = afrag_dot3(X1b[c], X1b[32 + c], X1b[64 + c],
                                  X1b + kb, X1b + 32 + kb, X1b + 64 + kb);
#pragma unroll
            for (int t = 0; t < 8; ++t) {
                v16bf b = load_bfrag(w110, c, 8, t, lane);
                accA[t] = wmma_bf16(aA, b, accA[t]);
                accB[t] = wmma_bf16(aB, b, accB[t]);
            }
        }
        act_store(accA, Aact[0], ln15, hf);
        act_store(accB, Aact[1], ln15, hf);
        lds_fence();

        // readout of scalar channel: z += s^T W2_000 s   (per sub-tile)
        quad_form<2, 1>(Aact[0], 128, w2000, lane, zacc[0]);
        quad_form<2, 1>(Aact[1], 128, w2000, lane, zacc[1]);

        // ===== Phase B: out1 (3 vector components), gate, readout =====
#pragma unroll 1
        for (int i = 0; i < 3; ++i) {
            const float* XiA = X1a + i * 32;
            const float* XiB = X1b + i * 32;
            v8f a1A[4], a1B[4];
#pragma unroll
            for (int t = 0; t < 4; ++t) { a1A[t] = v8f_zero(); a1B[t] = v8f_zero(); }
            for (int c = 0; c < 64; ++c) {       // G011, K = 2048
                v16bf aA = afrag_scale(X0a[c], XiA + kb);
                v16bf aB = afrag_scale(X0b[c], XiB + kb);
#pragma unroll
                for (int t = 0; t < 4; ++t) {
                    v16bf b = load_bfrag(w011, c, 4, t, lane);
                    a1A[t] = wmma_bf16(aA, b, a1A[t]);
                    a1B[t] = wmma_bf16(aB, b, a1B[t]);
                }
            }
            gate_store<4>(a1A, Aact[0], 32, Yb, ln15, hf);
            lds_fence();
            quad_form<4, 2>(Yb, 64, w2110, lane, zacc[0]);   // z += y1^T W2_110 y1
            lds_fence();
            gate_store<4>(a1B, Aact[1], 32, Yb, ln15, hf);
            lds_fence();
            quad_form<4, 2>(Yb, 64, w2110, lane, zacc[1]);
            lds_fence();
        }

        // ===== Phase B2: out2 (5 CG channels), gate, readout (per sub-tile) =====
        phase_out2(X1a, Aact[0], Yb, w112, w2220, lane, kb, ln15, hf, zacc[0]);
        phase_out2(X1b, Aact[1], Yb, w112, w2220, lane, kb, ln15, hf, zacc[1]);

        // ---- reduce per-lane partials (16 columns per row) ----
#pragma unroll
        for (int st = 0; st < 2; ++st)
#pragma unroll
            for (int r = 0; r < 8; ++r)
                atomicAdd(&zpart[s0 + st * 16 + hf * 8 + r], zacc[st][r]);
        __syncthreads();
        out[(size_t)blk * 64 + tid] = zpart[tid];
        __syncthreads();
    }
}

// ---------------------------------------------------------------------------
// Host launcher
// ---------------------------------------------------------------------------
extern "C" void kernel_launch(void* const* d_in, const int* in_sizes, int n_in,
                              void* d_out, int out_size, void* d_ws, size_t ws_size,
                              hipStream_t stream) {
    const float* x0 = (const float*)d_in[0];
    const float* x1 = (const float*)d_in[1];
    __bf16* wp = (__bf16*)d_ws;

    struct PackDesc { int idx; int K; int Nw; float scale; size_t off; };
    const PackDesc pd[7] = {
        {2, 4096, 128, 1.0f / 64.0f,  O_W000},   // w1_000 * 1/sqrt(64*64)
        {4, 1024, 128, 0.018042196f,  O_W110},   // w1_110 * 1/(32*sqrt(3))
        {3, 2048,  64, 0.022097087f,  O_W011},   // w1_011 * 1/sqrt(64*32)
        {5, 1024,  32, 1.0f / 32.0f,  O_W112},   // w1_112 * 1/sqrt(32*32)
        {6,   32,  32, 1.0f / 32.0f,  O_W2000},  // w2_000 * 1/32
        {7,   64,  64, 0.009021098f,  O_W2110},  // w2_110 * 1/(64*sqrt(3))
        {8,   32,  32, 0.013975425f,  O_W2220},  // w2_220 * 1/(32*sqrt(5))
    };
    for (int p = 0; p < 7; ++p) {
        long total = (long)pd[p].K * pd[p].Nw;
        int blocks = (int)((total + 255) / 256);
        pack_weight<<<blocks, 256, 0, stream>>>((const float*)d_in[pd[p].idx],
                                                wp + pd[p].off, pd[p].K, pd[p].Nw,
                                                pd[p].scale);
    }

    const int N = in_sizes[0] / 64;   // x0 is [N,64]
    const int nBlocks = N / 64;       // 64 samples per workgroup
    tp_gate_tp_kernel<<<nBlocks, 64, 0, stream>>>(x0, x1, wp, (float*)d_out, nBlocks);
}
